// HierarchicalGofLoss_2645699854493
// MI455X (gfx1250) — matrix operand order repaired and checked by reading, not compiled, not run on hardware
//
#include <hip/hip_runtime.h>
#include <hip/hip_bf16.h>
#include <math.h>

// ---------------------------------------------------------------------------
// HierarchicalGofLoss for MI455X (gfx1250, wave32, WMMA)
// Compile-time problem constants (match reference: D=512, K=32 children/family)
// ---------------------------------------------------------------------------
#define DIMS   512
#define KCH    32
#define RSTR   528          // padded bf16 LDS row stride (512 + 16) -> 1056 B

#define EPS_F        1e-8f
#define NORM_EPS_F   1e-12f
#define TARGET_COS_F 0.96592582628906828675f   // cos(15 deg)
#define DECAY_F      0.9f

typedef __attribute__((ext_vector_type(16))) __bf16 v16bf;
typedef __attribute__((ext_vector_type(8)))  __bf16 v8bf;
typedef __attribute__((ext_vector_type(4)))  __bf16 v4bf;
typedef __attribute__((ext_vector_type(8)))  float  v8f;

struct bfpair { v8bf lo, hi; };   // 32 bytes == v16bf

__device__ __forceinline__ float wave_sum32(float v) {
#pragma unroll
  for (int m = 16; m >= 1; m >>= 1) v += __shfl_xor(v, m, 32);
  return v;
}

__device__ __forceinline__ float dot4(const float4& a, const float4& b) {
  return a.x * b.x + a.y * b.y + a.z * b.z + a.w * b.w;
}

// --------------------------------------------------------------------------
// 0) zero the accumulators: acc[0]=orth, acc[1]=cos_sq, acc[2]=relu, acc[3]=brother
// --------------------------------------------------------------------------
__global__ void init_acc_kernel(float* acc) {
  if (threadIdx.x < 8) acc[threadIdx.x] = 0.0f;
}

// --------------------------------------------------------------------------
// 1) per-row L2 norms. One wave per row; float4 (global_load_b128) streaming.
// --------------------------------------------------------------------------
__global__ void row_norms_kernel(const float* __restrict__ E,
                                 float* __restrict__ norms, int n) {
  const int wv   = threadIdx.x >> 5;
  const int lane = threadIdx.x & 31;
  const int row  = blockIdx.x * 8 + wv;
  if (row >= n) return;
  const float4* p = (const float4*)(E + (size_t)row * DIMS);
  float ss = 0.0f;
#pragma unroll
  for (int t = 0; t < DIMS / 128; ++t) {
    float4 v = p[lane + 32 * t];
    ss += dot4(v, v);
  }
  ss = wave_sum32(ss);
  if (lane == 0) norms[row] = sqrtf(ss);
}

// --------------------------------------------------------------------------
// 2) parent-child cos angle MSE + length decay. One wave per pair.
// --------------------------------------------------------------------------
__global__ void parent_child_kernel(const float* __restrict__ E,
                                    const int* __restrict__ ci,
                                    const int* __restrict__ pi,
                                    const float* __restrict__ norms,
                                    float* __restrict__ acc, int npairs) {
  __shared__ float sA[8], sB[8];
  const int wv   = threadIdx.x >> 5;
  const int lane = threadIdx.x & 31;
  const int p    = blockIdx.x * 8 + wv;
  float a = 0.0f, b = 0.0f;
  if (p < npairs) {
    const int c = ci[p];
    const int q = pi[p];
    const float4* Ec = (const float4*)(E + (size_t)c * DIMS);
    const float4* Ep = (const float4*)(E + (size_t)q * DIMS);
    float dot = 0.0f;
#pragma unroll
    for (int t = 0; t < DIMS / 128; ++t)
      dot += dot4(Ec[lane + 32 * t], Ep[lane + 32 * t]);
    dot = wave_sum32(dot);
    const float nc  = norms[c];
    const float np_ = norms[q];
    const float cosv = dot / (fmaxf(nc, EPS_F) * fmaxf(np_, EPS_F));
    const float d = cosv - TARGET_COS_F;
    a = d * d;
    b = fmaxf(0.0f, DECAY_F * (np_ + EPS_F) - (nc + EPS_F));
  }
  if (lane == 0) { sA[wv] = a; sB[wv] = b; }
  __syncthreads();
  if (threadIdx.x == 0) {
    float sa = 0.0f, sb = 0.0f;
#pragma unroll
    for (int i = 0; i < 8; ++i) { sa += sA[i]; sb += sB[i]; }
    atomicAdd(&acc[1], sa);
    atomicAdd(&acc[2], sb);
  }
}

// --------------------------------------------------------------------------
// 3) global orthogonality among top labels: sum over i<j of (Tn_i . Tn_j)^2.
// --------------------------------------------------------------------------
__global__ void global_orth_kernel(const float* __restrict__ E,
                                   const int* __restrict__ top_idx,
                                   const float* __restrict__ norms,
                                   float* __restrict__ acc, int n_top) {
  __shared__ float sS[8];
  const int wv   = threadIdx.x >> 5;
  const int lane = threadIdx.x & 31;
  const int p    = blockIdx.x * 8 + wv;
  float s = 0.0f;
  if (p < n_top * n_top) {
    const int i = p / n_top;
    const int j = p % n_top;
    if (j > i) {
      const int ri = top_idx[i];
      const int rj = top_idx[j];
      const float4* Ei = (const float4*)(E + (size_t)ri * DIMS);
      const float4* Ej = (const float4*)(E + (size_t)rj * DIMS);
      float dot = 0.0f;
#pragma unroll
      for (int t = 0; t < DIMS / 128; ++t)
        dot += dot4(Ei[lane + 32 * t], Ej[lane + 32 * t]);
      dot = wave_sum32(dot);
      const float sim = dot / (fmaxf(norms[ri], NORM_EPS_F) *
                               fmaxf(norms[rj], NORM_EPS_F));
      s = sim * sim;
    }
  }
  if (lane == 0) sS[wv] = s;
  __syncthreads();
  if (threadIdx.x == 0) {
    float ss = 0.0f;
#pragma unroll
    for (int i = 0; i < 8; ++i) ss += sS[i];
    atomicAdd(&acc[0], ss);
  }
}

// --------------------------------------------------------------------------
// 4) brother residual decorrelation. One workgroup (128 thr = 4 waves) per
//    family. Residual rows normalized -> bf16 in LDS -> 2x2 tiled 32x32 Gram
//    via v_wmma_f32_16x16x32_bf16 (16 K-steps of 32 each).
// --------------------------------------------------------------------------
__global__ void __launch_bounds__(128)
brother_kernel(const float* __restrict__ E, const float* __restrict__ norms,
               const int* __restrict__ pA, const int* __restrict__ cA,
               const int* __restrict__ pB, const int* __restrict__ cB,
               int nA, float* __restrict__ acc) {
  __shared__ __attribute__((aligned(16))) float  sP[DIMS];
  __shared__ __attribute__((aligned(16))) __bf16 sRN[KCH * RSTR];
  __shared__ float  sPart[4];
  __shared__ float  sP2;
  __shared__ int    sSmall;

  const int f    = blockIdx.x;
  const int wv   = threadIdx.x >> 5;   // 0..3
  const int lane = threadIdx.x & 31;

  const int  parent   = (f < nA) ? pA[f] : pB[f - nA];
  const int* children = (f < nA) ? (cA + (size_t)f * KCH)
                                 : (cB + (size_t)(f - nA) * KCH);

  // stage parent row in LDS (128 lanes x one float4 = 512 floats)
  {
    const float4* Prow4 = (const float4*)(E + (size_t)parent * DIMS);
    ((float4*)sP)[threadIdx.x] = Prow4[threadIdx.x];
  }
  if (threadIdx.x == 0) {
    const float nrm = norms[parent];
    const float p2  = nrm * nrm;
    sSmall = (p2 < 1e-8f) ? 1 : 0;
    sP2    = sSmall ? 1.0f : p2;
  }
  __syncthreads();

  // residuals: each wave handles 8 children, float4 global streaming
  const float p2safe = sP2;
  const float4* sP4  = (const float4*)sP;
  for (int k = wv; k < KCH; k += 4) {
    const float4* Crow4 = (const float4*)(E + (size_t)children[k] * DIMS);
    float4 cv[DIMS / 128], pv[DIMS / 128];
    float dot = 0.0f;
#pragma unroll
    for (int t = 0; t < DIMS / 128; ++t) {
      cv[t] = Crow4[lane + 32 * t];
      pv[t] = sP4[lane + 32 * t];
      dot += dot4(cv[t], pv[t]);
    }
    dot = wave_sum32(dot);
    const float coef = dot / p2safe;
    float ss = 0.0f;
#pragma unroll
    for (int t = 0; t < DIMS / 128; ++t) {
      cv[t].x -= coef * pv[t].x;
      cv[t].y -= coef * pv[t].y;
      cv[t].z -= coef * pv[t].z;
      cv[t].w -= coef * pv[t].w;
      ss += dot4(cv[t], cv[t]);
    }
    ss = wave_sum32(ss);
    const float inv = 1.0f / fmaxf(sqrtf(ss), NORM_EPS_F);
#pragma unroll
    for (int t = 0; t < DIMS / 128; ++t) {
      v4bf o;
      o[0] = (__bf16)(cv[t].x * inv);
      o[1] = (__bf16)(cv[t].y * inv);
      o[2] = (__bf16)(cv[t].z * inv);
      o[3] = (__bf16)(cv[t].w * inv);
      // element index 4*(lane + 32*t); byte offset 8-aligned -> ds_store_b64
      *(v4bf*)(&sRN[(size_t)k * RSTR + 4 * (lane + 32 * t)]) = o;
    }
  }
  __syncthreads();

  // Gram G = rn @ rn^T, 2x2 tiles of 16x16; wave wv owns tile (ti,tj).
  // Strict-upper mask (N > M) zeroes the redundant (1,0) tile automatically.
  const int  ti  = wv >> 1;
  const int  tj  = wv & 1;
  const int  l15 = lane & 15;
  const bool hi  = lane >= 16;

  // A-matrix 16x32 bf16 layout (ISA 7.12.2): per lane, two contiguous 8-half
  // runs: lanes 0-15 at K=kb+0..7 / kb+16..23; lanes 16-31 at +8 / +24.
  // B-matrix 32x16: lane = column N; same K runs as A (kb+0..7 / kb+16..23; +8 for lanes>=16).
  const int koffA = hi ? 8 : 0;
  const int koffB = hi ? 8 : 0;
  const __bf16* arow = &sRN[(size_t)(ti * 16 + l15) * RSTR];
  const __bf16* brow = &sRN[(size_t)(tj * 16 + l15) * RSTR];

  v8f gacc = {};
#pragma unroll
  for (int s = 0; s < DIMS / 32; ++s) {
    const int kb = s * 32;
    const __bf16* ap = arow + kb + koffA;
    const __bf16* bp = brow + kb + koffB;
    bfpair at;                       // two aligned ds_load_b128 each
    at.lo = *(const v8bf*)(ap);
    at.hi = *(const v8bf*)(ap + 16);
    bfpair bt;
    bt.lo = *(const v8bf*)(bp);
    bt.hi = *(const v8bf*)(bp + 16);
    const v16bf af  = __builtin_bit_cast(v16bf, at);
    const v16bf bfv = __builtin_bit_cast(v16bf, bt);
    gacc = __builtin_amdgcn_wmma_f32_16x16x32_bf16(
        false, af, false, bfv, (short)0, gacc, false, false);
  }

  // C/D layout: VGPR r -> M = r (lanes 0-15) / r+8 (lanes 16-31), N = lane&15
  float local = 0.0f;
#pragma unroll
  for (int r = 0; r < 8; ++r) {
    const int M = ti * 16 + (hi ? r + 8 : r);
    const int N = tj * 16 + l15;
    if (N > M) local += fabsf(gacc[r]);
  }
  local = wave_sum32(local);
  if (lane == 0) sPart[wv] = local;
  __syncthreads();
  if (threadIdx.x == 0 && !sSmall) {
    const float fam = (sPart[0] + sPart[1] + sPart[2] + sPart[3]) *
                      (2.0f / (float)(KCH * (KCH - 1)));
    atomicAdd(&acc[3], fam);
  }
}

// --------------------------------------------------------------------------
// 5) combine
// --------------------------------------------------------------------------
__global__ void finalize_kernel(const float* __restrict__ acc,
                                float* __restrict__ out,
                                float n_top, float npairs, float nfam) {
  if (threadIdx.x == 0 && blockIdx.x == 0) {
    const float go = acc[0] * 2.0f / (n_top * (n_top - 1.0f));
    const float pc = acc[1] / npairs;
    const float ld = acc[2] / npairs;
    const float br = acc[3] / nfam;
    out[0] = go + pc + ld + br;
  }
}

// --------------------------------------------------------------------------
extern "C" void kernel_launch(void* const* d_in, const int* in_sizes, int n_in,
                              void* d_out, int out_size, void* d_ws, size_t ws_size,
                              hipStream_t stream) {
  const float* E   = (const float*)d_in[0];
  const int* ci    = (const int*)d_in[1];
  const int* pi    = (const int*)d_in[2];
  const int* top   = (const int*)d_in[3];
  const int* pA    = (const int*)d_in[4];
  const int* cA    = (const int*)d_in[5];
  const int* pB    = (const int*)d_in[6];
  const int* cB    = (const int*)d_in[7];

  const int npairs = in_sizes[1];             // 135168
  const int n_top  = in_sizes[3];             // 128
  const int nA     = in_sizes[4];             // 128 families
  const int nB     = in_sizes[6];             // 4096 families
  const int n      = n_top + npairs;          // total labels (135296)
  const int nfam   = nA + nB;

  float* acc   = (float*)d_ws;                // 8 floats
  float* norms = acc + 8;                     // n floats

  init_acc_kernel<<<1, 32, 0, stream>>>(acc);

  row_norms_kernel<<<(n + 7) / 8, 256, 0, stream>>>(E, norms, n);

  parent_child_kernel<<<(npairs + 7) / 8, 256, 0, stream>>>(
      E, ci, pi, norms, acc, npairs);

  global_orth_kernel<<<(n_top * n_top + 7) / 8, 256, 0, stream>>>(
      E, top, norms, acc, n_top);

  brother_kernel<<<nfam, 128, 0, stream>>>(E, norms, pA, cA, pB, cB, nA, acc);

  finalize_kernel<<<1, 32, 0, stream>>>(acc, (float*)d_out,
                                        (float)n_top, (float)npairs,
                                        (float)nfam);
}
